// STransformer_72593537237172
// MI455X (gfx1250) — compile-verified, matrix operand order checked by reading, not run on hardware
//
#include <hip/hip_runtime.h>

// ---------------- problem constants ----------------
constexpr int kB = 2, kN = 512, kR = 64, kT = 128, kE = 64, kH = 8, kD = 8, kFE = 4;
constexpr size_t SZ_BNTE = (size_t)kB * kN * kT * kE;   // 8388608
constexpr size_t SZ_BRTE = (size_t)kB * kR * kT * kE;   // 1048576
constexpr size_t SZ_ASR  = (size_t)kB * kR * kN * kT * kH; // 67108864
constexpr size_t SZ_ARR  = (size_t)kB * kR * kR * kT * kH; // 8388608
constexpr size_t SZ_ARS  = (size_t)kB * kN * kR * kT * kH; // 67108864
// d_out layout: out | asr | arr | ars | Hh
constexpr size_t OFF_ASR = SZ_BNTE;
constexpr size_t OFF_ARR = OFF_ASR + SZ_ASR;
constexpr size_t OFF_ARS = OFF_ARR + SZ_ARR;
constexpr size_t OFF_HH  = OFF_ARS + SZ_ARS;

typedef __attribute__((ext_vector_type(16))) __bf16 v16bf;
typedef __attribute__((ext_vector_type(8)))  __bf16 v8bf;
typedef __attribute__((ext_vector_type(8)))  float  v8f;
typedef __attribute__((ext_vector_type(4)))  float  v4f;

// ---------------- WMMA helpers ----------------
__device__ __forceinline__ v8f wmma_bf16(v16bf a, v16bf b, v8f c) {
  // D(f32 16x16) = A(bf16 16x32) * B(bf16 32x16) + C
  return __builtin_amdgcn_wmma_f32_16x16x32_bf16(false, a, false, b, (short)0, c, false, false);
}
__device__ __forceinline__ v8f vzero8() {
  v8f z;
#pragma unroll
  for (int i = 0; i < 8; ++i) z[i] = 0.f;
  return z;
}
__device__ __forceinline__ v16bf bzero16() {
  v16bf z;
#pragma unroll
  for (int i = 0; i < 16; ++i) z[i] = (__bf16)0.f;
  return z;
}

// ---------------- CDNA5 async global->LDS copy (ASYNCcnt path) ----------------
// Copies 16B per lane from global memory into LDS without touching VGPR data.
__device__ __forceinline__ void async_copy16(const void* g, void* lds) {
  uint32_t l = (uint32_t)(uintptr_t)lds;           // addrspace(3) offset = low 32 bits
  uint64_t ga = (uint64_t)(uintptr_t)g;
  asm volatile("global_load_async_to_lds_b128 %0, %1, off"
               :: "v"(l), "v"(ga) : "memory");
}
__device__ __forceinline__ void async_wait0() {
  asm volatile("s_wait_asynccnt 0x0" ::: "memory");
}

// Load an A-matrix (16x32 bf16) tile from row-major bf16 LDS.
// A striping: lane<16 holds K {kbase+0..7, kbase+16..23}; lane>=16 holds {kbase+8..15, kbase+24..31}.
// Both map to two contiguous 8xbf16 (16B) chunks -> two ds_load_b128.
__device__ __forceinline__ v16bf lds_loadA(const __bf16* base, int row, int stride,
                                           int kbase, int lane) {
  const int khalf = (lane >> 4) & 1;
  const __bf16* p = base + (size_t)row * stride + kbase + 8 * khalf;
  v8bf lo = *(const v8bf*)(p);
  v8bf hi = *(const v8bf*)(p + 16);
  v16bf r;
#pragma unroll
  for (int i = 0; i < 8; ++i) { r[i] = lo[i]; r[i + 8] = hi[i]; }
  return r;
}

// Gather a B-matrix (32x16 bf16) tile from a row-major f32 weight matrix W[k][col], ld = row stride.
__device__ __forceinline__ v16bf gatherB(const float* __restrict__ W, int ld, int col,
                                         int kbase, int lane) {
  const int khalf = (lane >> 4) & 1;
  v16bf r;
#pragma unroll
  for (int e = 0; e < 16; ++e) {
    int k = kbase + ((e < 8) ? e : e + 8) + 8 * khalf;
    r[e] = (__bf16)W[(size_t)k * ld + col];
  }
  return r;
}

// ---------------- small kernels ----------------
// ds[n,e] = sum_r D_S[n,r]*W_emb[r,e] + b_emb[e]
__global__ __launch_bounds__(256) void k_ds(const float* __restrict__ DS,
                                            const float* __restrict__ Wemb,
                                            const float* __restrict__ bemb,
                                            float* __restrict__ ds) {
  int id = blockIdx.x * 256 + threadIdx.x;
  int n = id >> 6, e = id & 63;
  float acc = bemb[e];
#pragma unroll 8
  for (int r = 0; r < kR; ++r) acc += DS[n * kR + r] * Wemb[r * kE + e];
  ds[id] = acc;
}

// q_plus = query + ds (broadcast over b,t); float4-vectorized
__global__ __launch_bounds__(256) void k_qplus(const float* __restrict__ query,
                                               const float* __restrict__ ds,
                                               float* __restrict__ qp) {
  size_t id = (size_t)blockIdx.x * 256 + threadIdx.x;  // float4 index
  int e4 = (int)(id & 15);
  int n  = (int)((id >> 11) & 511);
  v4f q = *(const v4f*)(query + id * 4);
  v4f d = *(const v4f*)(ds + (n * kE + e4 * 4));
  *(v4f*)(qp + id * 4) = q + d;
}

// x = layernorm(a + res)*g + be   (rows of 64, one wave per row)
__global__ __launch_bounds__(256) void k_add_ln(const float* __restrict__ a,
                                                const float* __restrict__ res,
                                                const float* __restrict__ g,
                                                const float* __restrict__ be,
                                                float* __restrict__ out) {
  size_t row = (size_t)blockIdx.x * 8 + (threadIdx.x >> 5);
  int lane = threadIdx.x & 31;
  const float* pa = a + row * kE;
  const float* pr = res + row * kE;
  float x0 = pa[lane] + pr[lane];
  float x1 = pa[lane + 32] + pr[lane + 32];
  float s = x0 + x1, s2 = x0 * x0 + x1 * x1;
#pragma unroll
  for (int o = 16; o > 0; o >>= 1) {
    s += __shfl_xor(s, o, 32);
    s2 += __shfl_xor(s2, o, 32);
  }
  float m = s * (1.f / 64.f);
  float v = s2 * (1.f / 64.f) - m * m;
  float rs = rsqrtf(v + 1e-5f);
  out[row * kE + lane]      = (x0 - m) * rs * g[lane] + be[lane];
  out[row * kE + lane + 32] = (x1 - m) * rs * g[lane + 32] + be[lane + 32];
}

// ---------------- attention: sensor->region (sr) ----------------
// per block: one (b,t) slice; wave w == head h.
__global__ __launch_bounds__(256) void k_attn_sr(
    const float* __restrict__ value, const float* __restrict__ key,
    const float* __restrict__ ds, const float* __restrict__ I_param,
    const int* __restrict__ adj_sr,
    const float* __restrict__ Wv, const float* __restrict__ Wk,
    const float* __restrict__ Wq, const float* __restrict__ Wo,
    const float* __restrict__ bo,
    float* __restrict__ Hh1, float* __restrict__ asr) {
  __shared__ __align__(16) float kraw[32][64];
  __shared__ __align__(16) float vraw[32][64];
  __shared__ float wk_s[64], wv_s[64], wq_s[64];
  __shared__ __align__(16) __bf16 att_st[8][64][32];
  __shared__ __align__(16) __bf16 out_st[64][64];

  const int b = blockIdx.x / kT, t = blockIdx.x % kT;
  const int lane = threadIdx.x & 31, w = threadIdx.x >> 5, h = w;
  const int lo16 = lane & 15, khalf = lane >> 4;

  if (threadIdx.x < 64) {
    wk_s[threadIdx.x] = Wk[threadIdx.x];
    wv_s[threadIdx.x] = Wv[threadIdx.x];
    wq_s[threadIdx.x] = Wq[threadIdx.x];
  }
  __syncthreads();

  // Q' = Iq @ Wq  (t-independent): A tiles, K padded 8->32
  v16bf Aq[4];
#pragma unroll
  for (int mt = 0; mt < 4; ++mt) {
    v16bf a = bzero16();
    if (khalf == 0) {
      int r = mt * 16 + lo16;
      float iq[8];
#pragma unroll
      for (int dd = 0; dd < 8; ++dd) iq[dd] = I_param[r * kE + h * 8 + dd];
#pragma unroll
      for (int d = 0; d < 8; ++d) {
        float acc = 0.f;
#pragma unroll
        for (int dd = 0; dd < 8; ++dd) acc += iq[dd] * wq_s[dd * 8 + d];
        a[d] = (__bf16)acc;
      }
    }
    Aq[mt] = a;
  }

  v8f outacc[4];
#pragma unroll
  for (int mt = 0; mt < 4; ++mt) outacc[mt] = vzero8();

  for (int chunk = 0; chunk < 16; ++chunk) {
    const int nbase = chunk * 32;
    __syncthreads();
    // value rows: pure copy -> CDNA5 async global->LDS DMA (ASYNCcnt)
#pragma unroll
    for (int i = 0; i < 2; ++i) {
      int j = threadIdx.x + i * 256;          // 512 float4 slots
      int rr = j >> 4, c4 = j & 15;
      int node = nbase + rr;
      const float* g = value + (((size_t)b * kN + node) * kT + t) * kE + c4 * 4;
      async_copy16(g, &vraw[rr][c4 * 4]);
    }
    // key rows need +ds: regular vectorized path
#pragma unroll
    for (int i = 0; i < 2; ++i) {
      int j = threadIdx.x + i * 256;
      int rr = j >> 4, c4 = j & 15;
      int node = nbase + rr;
      v4f kx = *(const v4f*)(key + (((size_t)b * kN + node) * kT + t) * kE + c4 * 4);
      v4f dx = *(const v4f*)(ds + node * kE + c4 * 4);
      *(v4f*)(&kraw[rr][c4 * 4]) = kx + dx;
    }
    async_wait0();
    __syncthreads();

    // energy B tiles: K'^T, K padded 8->32
    v16bf Bk[2];
#pragma unroll
    for (int nt = 0; nt < 2; ++nt) {
      v16bf bb = bzero16();
      if (khalf == 0) {
        int nl = nt * 16 + lo16;
#pragma unroll
        for (int d = 0; d < 8; ++d) {
          float acc = 0.f;
#pragma unroll
          for (int dd = 0; dd < 8; ++dd) acc += kraw[nl][h * 8 + dd] * wk_s[dd * 8 + d];
          bb[d] = (__bf16)acc;
        }
      }
      Bk[nt] = bb;
    }

    v8f en[4][2];
#pragma unroll
    for (int mt = 0; mt < 4; ++mt)
#pragma unroll
      for (int nt = 0; nt < 2; ++nt) en[mt][nt] = wmma_bf16(Aq[mt], Bk[nt], vzero8());

    // mask, scale, clip, exp
#pragma unroll
    for (int mt = 0; mt < 4; ++mt)
#pragma unroll
      for (int nt = 0; nt < 2; ++nt)
#pragma unroll
        for (int v = 0; v < 8; ++v) {
          int r = mt * 16 + v + 8 * khalf;
          int n = nbase + nt * 16 + lo16;
          float m = (float)adj_sr[n * kR + r];
          float x = en[mt][nt][v] * m - 1.0e8f * (1.f - m);
          x = fminf(5.f, fmaxf(-5.f, x * 0.125f));
          en[mt][nt][v] = __expf(x);
        }

    // softmax over q (axis=1): column sums over 64 rows, then normalize + emit
#pragma unroll
    for (int nt = 0; nt < 2; ++nt) {
      float s = 0.f;
#pragma unroll
      for (int mt = 0; mt < 4; ++mt)
#pragma unroll
        for (int v = 0; v < 8; ++v) s += en[mt][nt][v];
      s += __shfl_xor(s, 16, 32);
      float inv = 1.f / s;
      int n = nbase + nt * 16 + lo16;
#pragma unroll
      for (int mt = 0; mt < 4; ++mt)
#pragma unroll
        for (int v = 0; v < 8; ++v) {
          int r = mt * 16 + v + 8 * khalf;
          float at = en[mt][nt][v] * inv;
          asr[((((size_t)b * kR + r) * kN + n) * kT + t) * kH + h] = at;
          att_st[w][r][nt * 16 + lo16] = (__bf16)at;
        }
    }

    // out += att_chunk @ V'_chunk  (K = 32 nodes)
    v16bf Bv;
    {
      int d = lo16;
#pragma unroll
      for (int e = 0; e < 16; ++e) {
        float val = 0.f;
        if (d < 8) {
          int nl = ((e < 8) ? e : e + 8) + 8 * khalf;
#pragma unroll
          for (int dd = 0; dd < 8; ++dd) val += vraw[nl][h * 8 + dd] * wv_s[dd * 8 + d];
        }
        Bv[e] = (__bf16)val;
      }
    }
#pragma unroll
    for (int mt = 0; mt < 4; ++mt)
      outacc[mt] = wmma_bf16(lds_loadA(&att_st[w][0][0], mt * 16 + lo16, 32, 0, lane),
                             Bv, outacc[mt]);
  }

  // concat heads into LDS
  __syncthreads();
  if (lo16 < 8) {
#pragma unroll
    for (int mt = 0; mt < 4; ++mt)
#pragma unroll
      for (int v = 0; v < 8; ++v)
        out_st[mt * 16 + v + 8 * khalf][h * 8 + lo16] = (__bf16)outacc[mt][v];
  }
  __syncthreads();

  // Wo projection + bias -> Hh1 ; 16 tiles / 8 waves
#pragma unroll
  for (int i = 0; i < 2; ++i) {
    int tile = w * 2 + i;
    int mt = tile >> 2, nt = tile & 3;
    int col = nt * 16 + lo16;
    v8f acc = vzero8();
#pragma unroll
    for (int s = 0; s < 2; ++s)
      acc = wmma_bf16(lds_loadA(&out_st[0][0], mt * 16 + lo16, 64, 32 * s, lane),
                      gatherB(Wo, kE, col, 32 * s, lane), acc);
    float bias = bo[col];
#pragma unroll
    for (int v = 0; v < 8; ++v) {
      int r = mt * 16 + v + 8 * khalf;
      Hh1[(((size_t)b * kR + r) * kT + t) * kE + col] = acc[v] + bias;
    }
  }
}

// ---------------- attention: region->region (rr) ----------------
__global__ __launch_bounds__(256) void k_attn_rr(
    const float* __restrict__ Hh1, const int* __restrict__ adj_r,
    const float* __restrict__ Wv, const float* __restrict__ Wk,
    const float* __restrict__ Wq, const float* __restrict__ Wo,
    const float* __restrict__ bo,
    float* __restrict__ Hh2, float* __restrict__ arr, float* __restrict__ HhOut) {
  __shared__ __align__(16) float raw[64][64];
  __shared__ float wk_s[64], wv_s[64], wq_s[64];
  __shared__ __align__(16) __bf16 att_st[8][64][32];
  __shared__ __align__(16) __bf16 out_st[64][64];

  const int b = blockIdx.x / kT, t = blockIdx.x % kT;
  const int lane = threadIdx.x & 31, w = threadIdx.x >> 5, h = w;
  const int lo16 = lane & 15, khalf = lane >> 4;

  if (threadIdx.x < 64) {
    wk_s[threadIdx.x] = Wk[threadIdx.x];
    wv_s[threadIdx.x] = Wv[threadIdx.x];
    wq_s[threadIdx.x] = Wq[threadIdx.x];
  }
  // stage Hh1 (b,:,t,:) via async DMA: 64x64 f32 = 1024 float4
#pragma unroll
  for (int i = 0; i < 4; ++i) {
    int j = threadIdx.x + i * 256;
    int rr = j >> 4, c4 = j & 15;
    const float* g = Hh1 + (((size_t)b * kR + rr) * kT + t) * kE + c4 * 4;
    async_copy16(g, &raw[rr][c4 * 4]);
  }
  async_wait0();
  __syncthreads();

  v16bf Aq[4];
#pragma unroll
  for (int mt = 0; mt < 4; ++mt) {
    v16bf a = bzero16();
    if (khalf == 0) {
      int r = mt * 16 + lo16;
#pragma unroll
      for (int d = 0; d < 8; ++d) {
        float acc = 0.f;
#pragma unroll
        for (int dd = 0; dd < 8; ++dd) acc += raw[r][h * 8 + dd] * wq_s[dd * 8 + d];
        a[d] = (__bf16)acc;
      }
    }
    Aq[mt] = a;
  }

  v8f outacc[4];
#pragma unroll
  for (int mt = 0; mt < 4; ++mt) outacc[mt] = vzero8();

#pragma unroll
  for (int chunk = 0; chunk < 2; ++chunk) {
    const int nbase = chunk * 32;
    v16bf Bk[2];
#pragma unroll
    for (int nt = 0; nt < 2; ++nt) {
      v16bf bb = bzero16();
      if (khalf == 0) {
        int nl = nbase + nt * 16 + lo16;
#pragma unroll
        for (int d = 0; d < 8; ++d) {
          float acc = 0.f;
#pragma unroll
          for (int dd = 0; dd < 8; ++dd) acc += raw[nl][h * 8 + dd] * wk_s[dd * 8 + d];
          bb[d] = (__bf16)acc;
        }
      }
      Bk[nt] = bb;
    }
    v8f en[4][2];
#pragma unroll
    for (int mt = 0; mt < 4; ++mt)
#pragma unroll
      for (int nt = 0; nt < 2; ++nt) en[mt][nt] = wmma_bf16(Aq[mt], Bk[nt], vzero8());

#pragma unroll
    for (int mt = 0; mt < 4; ++mt)
#pragma unroll
      for (int nt = 0; nt < 2; ++nt)
#pragma unroll
        for (int v = 0; v < 8; ++v) {
          int r = mt * 16 + v + 8 * khalf;
          int n = nbase + nt * 16 + lo16;
          float m = (float)adj_r[r * kR + n];
          float x = en[mt][nt][v] * m - 1.0e8f * (1.f - m);
          x = fminf(5.f, fmaxf(-5.f, x * 0.125f));
          en[mt][nt][v] = __expf(x);
        }

#pragma unroll
    for (int nt = 0; nt < 2; ++nt) {
      float s = 0.f;
#pragma unroll
      for (int mt = 0; mt < 4; ++mt)
#pragma unroll
        for (int v = 0; v < 8; ++v) s += en[mt][nt][v];
      s += __shfl_xor(s, 16, 32);
      float inv = 1.f / s;
      int n = nbase + nt * 16 + lo16;
#pragma unroll
      for (int mt = 0; mt < 4; ++mt)
#pragma unroll
        for (int v = 0; v < 8; ++v) {
          int r = mt * 16 + v + 8 * khalf;
          float at = en[mt][nt][v] * inv;
          arr[((((size_t)b * kR + r) * kR + n) * kT + t) * kH + h] = at;
          att_st[w][r][nt * 16 + lo16] = (__bf16)at;
        }
    }

    v16bf Bv;
    {
      int d = lo16;
#pragma unroll
      for (int e = 0; e < 16; ++e) {
        float val = 0.f;
        if (d < 8) {
          int nl = nbase + ((e < 8) ? e : e + 8) + 8 * khalf;
#pragma unroll
          for (int dd = 0; dd < 8; ++dd) val += raw[nl][h * 8 + dd] * wv_s[dd * 8 + d];
        }
        Bv[e] = (__bf16)val;
      }
    }
#pragma unroll
    for (int mt = 0; mt < 4; ++mt)
      outacc[mt] = wmma_bf16(lds_loadA(&att_st[w][0][0], mt * 16 + lo16, 32, 0, lane),
                             Bv, outacc[mt]);
  }

  __syncthreads();
  if (lo16 < 8) {
#pragma unroll
    for (int mt = 0; mt < 4; ++mt)
#pragma unroll
      for (int v = 0; v < 8; ++v)
        out_st[mt * 16 + v + 8 * khalf][h * 8 + lo16] = (__bf16)outacc[mt][v];
  }
  __syncthreads();

#pragma unroll
  for (int i = 0; i < 2; ++i) {
    int tile = w * 2 + i;
    int mt = tile >> 2, nt = tile & 3;
    int col = nt * 16 + lo16;
    v8f acc = vzero8();
#pragma unroll
    for (int s = 0; s < 2; ++s)
      acc = wmma_bf16(lds_loadA(&out_st[0][0], mt * 16 + lo16, 64, 32 * s, lane),
                      gatherB(Wo, kE, col, 32 * s, lane), acc);
    float bias = bo[col];
#pragma unroll
    for (int v = 0; v < 8; ++v) {
      int r = mt * 16 + v + 8 * khalf;
      size_t off = (((size_t)b * kR + r) * kT + t) * kE + col;
      float val = acc[v] + bias;
      Hh2[off] = val;
      HhOut[off] = val;  // 5th tuple output
    }
  }
}

// ---------------- attention: region->sensor (rs) ----------------
// softmax axis=1 spans all 512 queries -> two passes (colsum, then normalize)
__global__ __launch_bounds__(256) void k_attn_rs(
    const float* __restrict__ qp, const float* __restrict__ Hh2,
    const float* __restrict__ Wv, const float* __restrict__ Wk,
    const float* __restrict__ Wq, const float* __restrict__ Wo,
    const float* __restrict__ bo,
    float* __restrict__ attn, float* __restrict__ ars) {
  __shared__ __align__(16) float raw[64][64];
  __shared__ float wk_s[64], wv_s[64], wq_s[64];
  __shared__ __align__(16) __bf16 att_st[8][64][32];
  __shared__ __align__(16) __bf16 out_st[64][64];

  const int b = blockIdx.x / kT, t = blockIdx.x % kT;
  const int lane = threadIdx.x & 31, w = threadIdx.x >> 5, h = w;
  const int lo16 = lane & 15, khalf = lane >> 4;

  if (threadIdx.x < 64) {
    wk_s[threadIdx.x] = Wk[threadIdx.x];
    wv_s[threadIdx.x] = Wv[threadIdx.x];
    wq_s[threadIdx.x] = Wq[threadIdx.x];
  }
  // stage Hh2 (b,:,t,:) via async DMA
#pragma unroll
  for (int i = 0; i < 4; ++i) {
    int j = threadIdx.x + i * 256;
    int rr = j >> 4, c4 = j & 15;
    const float* g = Hh2 + (((size_t)b * kR + rr) * kT + t) * kE + c4 * 4;
    async_copy16(g, &raw[rr][c4 * 4]);
  }
  async_wait0();
  __syncthreads();

  // K' (4 B-tiles over 64 region keys) and V' (2 K=32 chunks)
  v16bf Bk[4];
#pragma unroll
  for (int nt = 0; nt < 4; ++nt) {
    v16bf bb = bzero16();
    if (khalf == 0) {
      int nl = nt * 16 + lo16;
#pragma unroll
      for (int d = 0; d < 8; ++d) {
        float acc = 0.f;
#pragma unroll
        for (int dd = 0; dd < 8; ++dd) acc += raw[nl][h * 8 + dd] * wk_s[dd * 8 + d];
        bb[d] = (__bf16)acc;
      }
    }
    Bk[nt] = bb;
  }
  v16bf Bv[2];
#pragma unroll
  for (int kc = 0; kc < 2; ++kc) {
    int d = lo16;
#pragma unroll
    for (int e = 0; e < 16; ++e) {
      float val = 0.f;
      if (d < 8) {
        int nl = kc * 32 + ((e < 8) ? e : e + 8) + 8 * khalf;
#pragma unroll
        for (int dd = 0; dd < 8; ++dd) val += raw[nl][h * 8 + dd] * wv_s[dd * 8 + d];
      }
      Bv[kc][e] = (__bf16)val;
    }
  }

  // pass 1: column sums of exp(clip(energy/8)) over all 512 queries
  float csum[4] = {0.f, 0.f, 0.f, 0.f};
  for (int mc = 0; mc < 8; ++mc) {
    v16bf Aq[4];
#pragma unroll
    for (int mt = 0; mt < 4; ++mt) {
      v16bf a = bzero16();
      if (khalf == 0) {
        int q = mc * 64 + mt * 16 + lo16;
        float qr[8];
        const float* qptr = qp + (((size_t)b * kN + q) * kT + t) * kE + h * 8;
#pragma unroll
        for (int dd = 0; dd < 8; ++dd) qr[dd] = qptr[dd];
#pragma unroll
        for (int d = 0; d < 8; ++d) {
          float acc = 0.f;
#pragma unroll
          for (int dd = 0; dd < 8; ++dd) acc += qr[dd] * wq_s[dd * 8 + d];
          a[d] = (__bf16)acc;
        }
      }
      Aq[mt] = a;
    }
#pragma unroll
    for (int nt = 0; nt < 4; ++nt) {
#pragma unroll
      for (int mt = 0; mt < 4; ++mt) {
        v8f en = wmma_bf16(Aq[mt], Bk[nt], vzero8());
#pragma unroll
        for (int v = 0; v < 8; ++v) {
          float x = fminf(5.f, fmaxf(-5.f, en[v] * 0.125f));
          csum[nt] += __expf(x);
        }
      }
    }
  }
#pragma unroll
  for (int nt = 0; nt < 4; ++nt) csum[nt] += __shfl_xor(csum[nt], 16, 32);
  float cinv[4];
#pragma unroll
  for (int nt = 0; nt < 4; ++nt) cinv[nt] = 1.f / csum[nt];

  // pass 2: normalize, emit ars, out = att@V', Wo projection per 64-query chunk
  for (int mc = 0; mc < 8; ++mc) {
    v16bf Aq[4];
#pragma unroll
    for (int mt = 0; mt < 4; ++mt) {
      v16bf a = bzero16();
      if (khalf == 0) {
        int q = mc * 64 + mt * 16 + lo16;
        float qr[8];
        const float* qptr = qp + (((size_t)b * kN + q) * kT + t) * kE + h * 8;
#pragma unroll
        for (int dd = 0; dd < 8; ++dd) qr[dd] = qptr[dd];
#pragma unroll
        for (int d = 0; d < 8; ++d) {
          float acc = 0.f;
#pragma unroll
          for (int dd = 0; dd < 8; ++dd) acc += qr[dd] * wq_s[dd * 8 + d];
          a[d] = (__bf16)acc;
        }
      }
      Aq[mt] = a;
    }

    v8f outacc[4];
#pragma unroll
    for (int mt = 0; mt < 4; ++mt) outacc[mt] = vzero8();

#pragma unroll
    for (int kc = 0; kc < 2; ++kc) {
#pragma unroll
      for (int nt = 0; nt < 2; ++nt) {
        int gnt = kc * 2 + nt;
        int n = gnt * 16 + lo16;
#pragma unroll
        for (int mt = 0; mt < 4; ++mt) {
          v8f en = wmma_bf16(Aq[mt], Bk[gnt], vzero8());
#pragma unroll
          for (int v = 0; v < 8; ++v) {
            int q = mc * 64 + mt * 16 + v + 8 * khalf;
            float x = fminf(5.f, fmaxf(-5.f, en[v] * 0.125f));
            float at = __expf(x) * cinv[gnt];
            ars[((((size_t)b * kN + q) * kR + n) * kT + t) * kH + h] = at;
            att_st[w][mt * 16 + v + 8 * khalf][nt * 16 + lo16] = (__bf16)at;
          }
        }
      }
#pragma unroll
      for (int mt = 0; mt < 4; ++mt)
        outacc[mt] = wmma_bf16(lds_loadA(&att_st[w][0][0], mt * 16 + lo16, 32, 0, lane),
                               Bv[kc], outacc[mt]);
    }

    // concat heads for this query chunk + Wo projection
    __syncthreads();
    if (lo16 < 8) {
#pragma unroll
      for (int mt = 0; mt < 4; ++mt)
#pragma unroll
        for (int v = 0; v < 8; ++v)
          out_st[mt * 16 + v + 8 * khalf][h * 8 + lo16] = (__bf16)outacc[mt][v];
    }
    __syncthreads();

#pragma unroll
    for (int i = 0; i < 2; ++i) {
      int tile = w * 2 + i;
      int mt = tile >> 2, nt = tile & 3;
      int col = nt * 16 + lo16;
      v8f acc = vzero8();
#pragma unroll
      for (int s = 0; s < 2; ++s)
        acc = wmma_bf16(lds_loadA(&out_st[0][0], mt * 16 + lo16, 64, 32 * s, lane),
                        gatherB(Wo, kE, col, 32 * s, lane), acc);
      float bias = bo[col];
#pragma unroll
      for (int v = 0; v < 8; ++v) {
        int q = mc * 64 + mt * 16 + v + 8 * khalf;
        attn[(((size_t)b * kN + q) * kT + t) * kE + col] = acc[v] + bias;
      }
    }
    __syncthreads();
  }
}

// ---------------- fused FFN + residual-LN + final projection ----------------
// per block: 64 rows; out = (LN(relu(x@W1+b1)@W2+b2 + x))@Wfs + bfs
__global__ __launch_bounds__(256) void k_ffn(
    const float* __restrict__ x, const float* __restrict__ W1,
    const float* __restrict__ b1, const float* __restrict__ W2,
    const float* __restrict__ b2, const float* __restrict__ g2,
    const float* __restrict__ be2, const float* __restrict__ Wfs,
    const float* __restrict__ bfs, float* __restrict__ out) {
  __shared__ __align__(16) float xf[64][64];     // x tile, then U=y+x (f32)
  __shared__ __align__(16) __bf16 xb[64][64];    // x bf16, then LN(U) bf16
  __shared__ __align__(16) __bf16 hb[64][256];   // hidden bf16

  const int lane = threadIdx.x & 31, w = threadIdx.x >> 5;
  const int lo16 = lane & 15, khalf = lane >> 4;
  const size_t rowbase = (size_t)blockIdx.x * 64;

  // stage x tile via async DMA, then mirror to bf16
#pragma unroll
  for (int i = 0; i < 4; ++i) {
    int j = threadIdx.x + i * 256;  // 1024 float4
    async_copy16(x + rowbase * kE + (size_t)j * 4, &xf[0][0] + j * 4);
  }
  async_wait0();
  __syncthreads();
  for (int i = threadIdx.x; i < 64 * 64; i += 256)
    (&xb[0][0])[i] = (__bf16)(&xf[0][0])[i];
  __syncthreads();

  // GEMM1: hidden = relu(x @ W1 + b1)   [64x64 @ 64x256]
#pragma unroll
  for (int j = 0; j < 2; ++j) {
    int nt = w * 2 + j;
    int col = nt * 16 + lo16;
    v16bf B0 = gatherB(W1, kFE * kE, col, 0, lane);
    v16bf B1 = gatherB(W1, kFE * kE, col, 32, lane);
    float bias = b1[col];
#pragma unroll
    for (int mt = 0; mt < 4; ++mt) {
      v8f acc = vzero8();
      acc = wmma_bf16(lds_loadA(&xb[0][0], mt * 16 + lo16, 64, 0, lane), B0, acc);
      acc = wmma_bf16(lds_loadA(&xb[0][0], mt * 16 + lo16, 64, 32, lane), B1, acc);
#pragma unroll
      for (int v = 0; v < 8; ++v) {
        int q = mt * 16 + v + 8 * khalf;
        hb[q][col] = (__bf16)fmaxf(acc[v] + bias, 0.f);
      }
    }
  }
  __syncthreads();

  // GEMM2: y = h @ W2 + b2, then U = y + x  (written over xf)
#pragma unroll
  for (int j = 0; j < 2; ++j) {
    int tile = w * 2 + j;
    int mt = tile >> 2, nt = tile & 3;
    int col = nt * 16 + lo16;
    v8f acc = vzero8();
#pragma unroll
    for (int s = 0; s < 8; ++s)
      acc = wmma_bf16(lds_loadA(&hb[0][0], mt * 16 + lo16, 256, 32 * s, lane),
                      gatherB(W2, kE, col, 32 * s, lane), acc);
    float bias = b2[col];
#pragma unroll
    for (int v = 0; v < 8; ++v) {
      int q = mt * 16 + v + 8 * khalf;
      xf[q][col] = acc[v] + bias + xf[q][col];
    }
  }
  __syncthreads();

  // LayerNorm rows of U -> xb (bf16)
#pragma unroll
  for (int rr = 0; rr < 8; ++rr) {
    int q = w * 8 + rr;
    float x0 = xf[q][lane], x1 = xf[q][lane + 32];
    float s = x0 + x1, s2 = x0 * x0 + x1 * x1;
#pragma unroll
    for (int o = 16; o > 0; o >>= 1) {
      s += __shfl_xor(s, o, 32);
      s2 += __shfl_xor(s2, o, 32);
    }
    float m = s * (1.f / 64.f);
    float var = s2 * (1.f / 64.f) - m * m;
    float rs = rsqrtf(var + 1e-5f);
    xb[q][lane]      = (__bf16)((x0 - m) * rs * g2[lane] + be2[lane]);
    xb[q][lane + 32] = (__bf16)((x1 - m) * rs * g2[lane + 32] + be2[lane + 32]);
  }
  __syncthreads();

  // GEMM3: out = LN(U) @ Wfs + bfs
#pragma unroll
  for (int j = 0; j < 2; ++j) {
    int tile = w * 2 + j;
    int mt = tile >> 2, nt = tile & 3;
    int col = nt * 16 + lo16;
    v8f acc = vzero8();
#pragma unroll
    for (int s = 0; s < 2; ++s)
      acc = wmma_bf16(lds_loadA(&xb[0][0], mt * 16 + lo16, 64, 32 * s, lane),
                      gatherB(Wfs, kE, col, 32 * s, lane), acc);
    float bias = bfs[col];
#pragma unroll
    for (int v = 0; v < 8; ++v) {
      int q = mt * 16 + v + 8 * khalf;
      out[(rowbase + q) * kE + col] = acc[v] + bias;
    }
  }
}

// ---------------- host launch ----------------
extern "C" void kernel_launch(void* const* d_in, const int* in_sizes, int n_in,
                              void* d_out, int out_size, void* d_ws, size_t ws_size,
                              hipStream_t stream) {
  const float* value  = (const float*)d_in[0];
  const float* key    = (const float*)d_in[1];
  const float* query  = (const float*)d_in[2];
  const int*   adj_sr = (const int*)d_in[3];
  const int*   adj_r  = (const int*)d_in[4];
  const float* D_S    = (const float*)d_in[5];
  const float* W_emb  = (const float*)d_in[6];
  const float* b_emb  = (const float*)d_in[7];
  const float* I_par  = (const float*)d_in[8];
  const float* g1     = (const float*)d_in[9];
  const float* beta1  = (const float*)d_in[10];
  const float* g2     = (const float*)d_in[11];
  const float* beta2  = (const float*)d_in[12];
  const float* W_ff1  = (const float*)d_in[13];
  const float* b_ff1  = (const float*)d_in[14];
  const float* W_ff2  = (const float*)d_in[15];
  const float* b_ff2  = (const float*)d_in[16];
  const float* W_fs   = (const float*)d_in[17];
  const float* b_fs   = (const float*)d_in[18];
  const float* Wv_sr = (const float*)d_in[19], *Wk_sr = (const float*)d_in[20];
  const float* Wq_sr = (const float*)d_in[21], *Wo_sr = (const float*)d_in[22];
  const float* bo_sr = (const float*)d_in[23];
  const float* Wv_rr = (const float*)d_in[24], *Wk_rr = (const float*)d_in[25];
  const float* Wq_rr = (const float*)d_in[26], *Wo_rr = (const float*)d_in[27];
  const float* bo_rr = (const float*)d_in[28];
  const float* Wv_rs = (const float*)d_in[29], *Wk_rs = (const float*)d_in[30];
  const float* Wq_rs = (const float*)d_in[31], *Wo_rs = (const float*)d_in[32];
  const float* bo_rs = (const float*)d_in[33];

  float* outp  = (float*)d_out;
  float* asr   = outp + OFF_ASR;
  float* arr   = outp + OFF_ARR;
  float* ars   = outp + OFF_ARS;
  float* HhOut = outp + OFF_HH;

  float* ws   = (float*)d_ws;
  float* ds   = ws;
  float* qp   = ds + (size_t)kN * kE;
  float* Hh1  = qp + SZ_BNTE;
  float* Hh2  = Hh1 + SZ_BRTE;
  float* attn = Hh2 + SZ_BRTE;
  float* xbuf = attn + SZ_BNTE;

  k_ds<<<(kN * kE) / 256, 256, 0, stream>>>(D_S, W_emb, b_emb, ds);
  k_qplus<<<(int)(SZ_BNTE / 4 / 256), 256, 0, stream>>>(query, ds, qp);
  k_attn_sr<<<kB * kT, 256, 0, stream>>>(value, key, ds, I_par, adj_sr,
                                         Wv_sr, Wk_sr, Wq_sr, Wo_sr, bo_sr, Hh1, asr);
  k_attn_rr<<<kB * kT, 256, 0, stream>>>(Hh1, adj_r, Wv_rr, Wk_rr, Wq_rr, Wo_rr, bo_rr,
                                         Hh2, arr, HhOut);
  k_attn_rs<<<kB * kT, 256, 0, stream>>>(qp, Hh2, Wv_rs, Wk_rs, Wq_rs, Wo_rs, bo_rs,
                                         attn, ars);
  k_add_ln<<<(int)(SZ_BNTE / kE / 8), 256, 0, stream>>>(attn, qp, g1, beta1, xbuf);
  k_ffn<<<(int)(SZ_BNTE / kE / 64), 256, 0, stream>>>(xbuf, W_ff1, b_ff1, W_ff2, b_ff2,
                                                      g2, beta2, W_fs, b_fs, outp);
}
